// InitialResidueEmbedding_71554155151991
// MI455X (gfx1250) — matrix operand order repaired and checked by reading, not compiled, not run on hardware
//
#include <hip/hip_runtime.h>
#include <hip/hip_bf16.h>

// ---------- problem constants (from reference) ----------
#define PAD   14
#define ACD   32
#define ATD   8
#define RCD   32
#define RID   32
#define IN0   624          // 14*32 + 14*8 + 32 + 32
#define KPAD  640          // IN0 padded to multiple of 32
#define NKC   (KPAD / 32)  // 20 K-chunks
#define OUT0  128
#define OUT1  32
#define FEAT  (OUT0 + OUT1 * 3)   // 224
#define MT    16           // residues per block (M tile)
// LDS A-tile row stride in dwords: 320 data dwords + 4 pad (conflict-free frag loads)
#define ASTR  324

typedef __attribute__((ext_vector_type(16))) __bf16 v16bf;
typedef __attribute__((ext_vector_type(8)))  float  v8f;

union FragU {
    int4  q[2];   // 8 dwords = 16 bf16
    v16bf v;
};

__device__ __forceinline__ unsigned bf16pack(float a, float b) {
    unsigned ua = __builtin_bit_cast(unsigned, a);
    unsigned ub = __builtin_bit_cast(unsigned, b);
    ua = (ua + 0x7FFFu + ((ua >> 16) & 1u)) >> 16;   // RNE float->bf16
    ub = (ub + 0x7FFFu + ((ub >> 16) & 1u)) >> 16;
    return (ub << 16) | (ua & 0xFFFFu);
}

// ---------------------------------------------------------------------------
// P1: repack W0 [624,128] fp32 row-major -> bf16 WMMA-B fragments, K padded
// to 640. Layout: frag(kc, nt)[lane][v]  (v = 0..7 dwords per lane).
// B (32x16 bf16), k order chosen here to match A: VGPR v, lanes 0-15 hold rows {K=2v,2v+1} at col=lane,
// lanes 16-31 hold rows {K=2v+16,2v+17} at col=lane-16.
// ---------------------------------------------------------------------------
__global__ void prep_w0_kernel(const float* __restrict__ W0,
                               unsigned* __restrict__ Bws) {
    const int kc   = blockIdx.x >> 3;       // 0..19
    const int nt   = blockIdx.x & 7;        // 0..7
    const int lane = threadIdx.x;           // 0..31
    const int n    = nt * 16 + (lane & 15);
    const int kbase = kc * 32 + ((lane & 16) ? 16 : 0);
    unsigned* dst = Bws + ((size_t)(kc * 8 + nt) * 32 + lane) * 8;
#pragma unroll
    for (int v = 0; v < 8; ++v) {
        const int k = kbase + 2 * v;
        float f0 = (k < IN0) ? W0[(size_t)k * OUT0 + n] : 0.0f;
        float f1 = (k < IN0) ? W0[(size_t)(k + 1) * OUT0 + n] : 0.0f;
        dst[v] = bf16pack(f0, f1);
    }
}

// ---------------------------------------------------------------------------
// P2: residue start offsets from contiguous-ascending residue_index_atomwise.
// ---------------------------------------------------------------------------
__global__ void prep_starts_kernel(const int* __restrict__ ridx,
                                   int* __restrict__ starts, int n_atoms) {
    int i = blockIdx.x * blockDim.x + threadIdx.x;
    if (i >= n_atoms) return;
    int r = ridx[i];
    if (i == 0 || ridx[i - 1] != r) starts[r] = i;
}

// ---------------------------------------------------------------------------
// MAIN: one block (128 threads = 4 waves) per 16-residue tile.
// ---------------------------------------------------------------------------
__global__ void __launch_bounds__(128)
residue_embed_kernel(const float* __restrict__ coords,
                     const int*   __restrict__ acode,
                     const int*   __restrict__ atype,
                     const int*   __restrict__ starts,
                     const int*   __restrict__ counts,
                     const int*   __restrict__ rcode,
                     const int*   __restrict__ rseq,
                     const float* __restrict__ base_coords,
                     const float* __restrict__ acode_tab,
                     const float* __restrict__ atype_tab,
                     const float* __restrict__ rcode_tab,
                     const float* __restrict__ ridx_tab,
                     const unsigned* __restrict__ Bws,
                     const float* __restrict__ W1,
                     float* __restrict__ out, int R) {
    __shared__ unsigned sA[MT * ASTR];          // 16 x 640 bf16 A-tile (padded rows)
    __shared__ float    sC[MT * PAD * 3];       // coords_pad
    __shared__ float    sW1[PAD * OUT1];
    __shared__ int      sStart[MT], sCount[MT], sRC[MT], sRI[MT];
    __shared__ int      sCode[MT * PAD], sType[MT * PAD];

    const int t    = threadIdx.x;
    const int tile = blockIdx.x;
    const size_t R3 = (size_t)R * 3;

    // phase 0: per-residue metadata
    if (t < MT) {
        int r = tile * MT + t;
        if (r < R) {
            sStart[t] = starts[r];
            sCount[t] = counts[r];
            sRC[t]    = rcode[r];
            sRI[t]    = rseq[r];
        } else {
            sStart[t] = 0; sCount[t] = 0; sRC[t] = -1; sRI[t] = -1;
        }
    }
    __syncthreads();

    // phase 1: gather codes/types, coords, W1
    for (int idx = t; idx < MT * PAD; idx += 128) {
        int rl = idx / PAD, m = idx % PAD;
        bool ok = m < sCount[rl];
        int a0 = ok ? acode[sStart[rl] + m] : -1;
        int a1 = ok ? atype[sStart[rl] + m] : -1;
        sCode[idx] = a0; sType[idx] = a1;
    }
    for (int idx = t; idx < MT * PAD * 3; idx += 128) {
        int rl = idx / (PAD * 3), rem = idx % (PAD * 3);
        int m = rem / 3, c = rem % 3;
        float v = 0.0f;
        if (m < sCount[rl]) v = coords[(size_t)(sStart[rl] + m) * 3 + c];
        sC[idx] = v;
    }
    for (int idx = t; idx < PAD * OUT1; idx += 128) sW1[idx] = W1[idx];
    __syncthreads();

    // phase 2: build bf16 A-tile (row-major [16][640], stride 324 dwords)
    {
        const int rl  = t >> 3;       // residue in tile
        const int sub = t & 7;
        const int cnt = sCount[rl];
        const int rc  = sRC[rl], ri = sRI[rl];
        for (int d = sub; d < ASTR; d += 8) {
            float f0 = 0.0f, f1 = 0.0f;
            if (d < 224) {                       // atom-code embeds: 14 x 32
                int m = d >> 4, j = d & 15;
                int c = sCode[rl * PAD + m];
                if (m < cnt && c >= 0) {
                    f0 = acode_tab[c * ACD + 2 * j];
                    f1 = acode_tab[c * ACD + 2 * j + 1];
                }
            } else if (d < 280) {                // atom-type embeds: 14 x 8
                int dd = d - 224, m = dd >> 2, j = dd & 3;
                int c = sType[rl * PAD + m];
                if (m < cnt && c >= 0) {
                    f0 = atype_tab[c * ATD + 2 * j];
                    f1 = atype_tab[c * ATD + 2 * j + 1];
                }
            } else if (d < 296) {                // residue-code embed: 32
                int j = d - 280;
                if (rc >= 0) {
                    f0 = rcode_tab[rc * RCD + 2 * j];
                    f1 = rcode_tab[rc * RCD + 2 * j + 1];
                }
            } else if (d < 312) {                // residue-index embed: 32
                int j = d - 296;
                if (ri >= 0) {
                    f0 = ridx_tab[ri * RID + 2 * j];
                    f1 = ridx_tab[ri * RID + 2 * j + 1];
                }
            }                                     // d in [312,324): zero K-pad
            sA[rl * ASTR + d] = bf16pack(f0, f1);
        }
    }
    __syncthreads();

    // phase 3: WMMA. 4 waves x 2 N-tiles = full 128 output channels.
    const int wave = t >> 5;
    const int lane = t & 31;
    const int nt0 = wave * 2, nt1 = nt0 + 1;
    v8f acc0 = {0.f,0.f,0.f,0.f,0.f,0.f,0.f,0.f};
    v8f acc1 = {0.f,0.f,0.f,0.f,0.f,0.f,0.f,0.f};

    const int arow = lane & 15;
    const int aoff = (lane >> 4) << 3;           // 0 or 8 dwords
#pragma unroll
    for (int kc = 0; kc < NKC; ++kc) {
        FragU a, b0, b1;
        const int4* pa = reinterpret_cast<const int4*>(&sA[arow * ASTR + kc * 16 + aoff]);
        a.q[0] = pa[0];                          // K pairs 0..3  (or 8..11)
        a.q[1] = pa[1];                          // K pairs 4..7  (or 12..15)
        const int4* pb0 = reinterpret_cast<const int4*>(
            Bws + ((size_t)(kc * 8 + nt0) * 32 + lane) * 8);
        b0.q[0] = pb0[0]; b0.q[1] = pb0[1];
        const int4* pb1 = reinterpret_cast<const int4*>(
            Bws + ((size_t)(kc * 8 + nt1) * 32 + lane) * 8);
        b1.q[0] = pb1[0]; b1.q[1] = pb1[1];
        acc0 = __builtin_amdgcn_wmma_f32_16x16x32_bf16(
            false, a.v, false, b0.v, (short)0, acc0, false, false);
        acc1 = __builtin_amdgcn_wmma_f32_16x16x32_bf16(
            false, a.v, false, b1.v, (short)0, acc1, false, false);
    }

    // y0 store: C layout => VGPR v: lanes 0-15 row M=v, lanes 16-31 row M=v+8
    const float s0 = 1.0f / sqrtf((float)IN0);
    const int rbase = (lane >> 4) << 3;
    const int col0 = nt0 * 16 + (lane & 15);
    const int col1 = nt1 * 16 + (lane & 15);
#pragma unroll
    for (int v = 0; v < 8; ++v) {
        int row = tile * MT + rbase + v;
        if (row < R) {
            out[R3 + (size_t)row * FEAT + col0] = acc0[v] * s0;
            out[R3 + (size_t)row * FEAT + col1] = acc1[v] * s0;
        }
    }

    // phase 4: 1e path  y1[r,n,c] = sum_m coords_pad[r,m,c]*W1[m,n] / sqrt(14)
    const float s1 = 1.0f / sqrtf((float)PAD);
    for (int idx = t; idx < MT * OUT1 * 3; idx += 128) {
        int rl = idx / (OUT1 * 3), o = idx % (OUT1 * 3);
        int n = o / 3, c = o % 3;
        int row = tile * MT + rl;
        if (row < R) {
            float s = 0.0f;
#pragma unroll
            for (int m = 0; m < PAD; ++m)
                s += sC[(rl * PAD + m) * 3 + c] * sW1[m * OUT1 + n];
            out[R3 + (size_t)row * FEAT + OUT0 + n * 3 + c] = s * s1;
        }
    }

    // base_coords passthrough
    for (int idx = t; idx < MT * 3; idx += 128) {
        size_t g = (size_t)tile * MT * 3 + idx;
        if (g < R3) out[g] = base_coords[g];
    }
}

// ---------------------------------------------------------------------------
extern "C" void kernel_launch(void* const* d_in, const int* in_sizes, int n_in,
                              void* d_out, int out_size, void* d_ws, size_t ws_size,
                              hipStream_t stream) {
    const float* coords = (const float*)d_in[0];
    const int*   acode  = (const int*)  d_in[1];
    const int*   atype  = (const int*)  d_in[2];
    const int*   ridx   = (const int*)  d_in[3];
    const int*   counts = (const int*)  d_in[4];
    const int*   rcode  = (const int*)  d_in[5];
    const int*   rseq   = (const int*)  d_in[6];
    const float* basec  = (const float*)d_in[7];
    const float* actab  = (const float*)d_in[8];
    const float* attab  = (const float*)d_in[9];
    const float* rctab  = (const float*)d_in[10];
    const float* ritab  = (const float*)d_in[11];
    const float* W0     = (const float*)d_in[12];
    const float* W1     = (const float*)d_in[13];
    float* out = (float*)d_out;

    const int n_atoms = in_sizes[3];
    const int R       = in_sizes[4];

    // workspace: [starts: R ints][align 256][B fragments: 20*8*32*8 dwords bf16-packed]
    int* starts = (int*)d_ws;
    size_t boff = ((size_t)R * sizeof(int) + 255) & ~(size_t)255;
    unsigned* Bws = (unsigned*)((char*)d_ws + boff);

    prep_w0_kernel<<<dim3(NKC * 8), dim3(32), 0, stream>>>(W0, Bws);
    prep_starts_kernel<<<dim3((n_atoms + 255) / 256), dim3(256), 0, stream>>>(
        ridx, starts, n_atoms);

    const int ntiles = (R + MT - 1) / MT;
    residue_embed_kernel<<<dim3(ntiles), dim3(128), 0, stream>>>(
        coords, acode, atype, starts, counts, rcode, rseq, basec,
        actab, attab, rctab, ritab, Bws, W1, out, R);
}
